// MultiheadAttentionEdge_62603443306949
// MI455X (gfx1250) — matrix operand order repaired and checked by reading, not compiled, not run on hardware
//
#include <hip/hip_runtime.h>

// ---------------------------------------------------------------------------
// MultiheadAttentionEdge on MI455X (gfx1250, wave32, WMMA f32_16x16x32_f16)
// N=2048 nodes, E=512, H=8 heads, hd=64, 10 edge types.
// Flash-style fused attention: never materializes the (H,N,N) score matrix.
// All GEMM operands pre-converted to f16 once; inner loops are software-
// pipelined f16 b128 fragment loads + v_wmma_f32_16x16x32_f16.
// Softmax row-sums are computed by an extra WMMA against an all-ones B.
// ---------------------------------------------------------------------------

#define N_NODES 2048
#define EDIM    512
#define HD      64
#define NH      8
#define NTYPES  10

typedef _Float16 v16h __attribute__((ext_vector_type(16)));
typedef _Float16 v8h  __attribute__((ext_vector_type(8)));
typedef _Float16 v4h  __attribute__((ext_vector_type(4)));
typedef float    v8f  __attribute__((ext_vector_type(8)));
typedef float    v4f  __attribute__((ext_vector_type(4)));

#define WMMA_F32_F16(a, b, c) \
  __builtin_amdgcn_wmma_f32_16x16x32_f16(false, (a), false, (b), (short)0, (c), false, false)

static __device__ __forceinline__ v16h join16(v8h lo, v8h hi) {
  return __builtin_shufflevector(lo, hi, 0, 1, 2, 3, 4, 5, 6, 7,
                                         8, 9, 10, 11, 12, 13, 14, 15);
}
// A-fragment (16x32 f16, MxK): lane holds row M=lane%16.
// lanes 0-15 hold K=[0..7],[16..23]; lanes 16-31 hold K=[8..15],[24..31].
// Caller passes p = rowptr + kbase + (lane>=16 ? 8 : 0); p is 16B aligned.
static __device__ __forceinline__ v16h afrag_h(const _Float16* p) {
  return join16(*(const v8h*)p, *(const v8h*)(p + 16));
}
// B-fragment (32x16 f16, KxN): lane holds column N=lane%16.
// lanes 0-15 hold K=[0..7],[16..23]; lanes 16-31 hold K=[8..15],[24..31].
// Caller passes p = colptr + kbase + (lane>=16 ? 8 : 0); p is 16B aligned.
static __device__ __forceinline__ v16h bfrag_h(const _Float16* p) {
  return join16(*(const v8h*)p, *(const v8h*)(p + 16));
}

// Max-reduction butterfly within each 16-lane half of the wave32 (masks 1..8
// never cross the half boundary) — matches the WMMA C layout where one
// matrix row spans exactly one 16-lane half.
static __device__ __forceinline__ float half_max(float v) {
#pragma unroll
  for (int mask = 1; mask < 16; mask <<= 1) v = fmaxf(v, __shfl_xor(v, mask, 32));
  return v;
}

// ---------------------------------------------------------------------------
// K0: fp32 -> f16 bulk convert (vectorized b128 in / b64 out). n4 = n/4.
// ---------------------------------------------------------------------------
__global__ __launch_bounds__(256) void cvt_f16_kernel(
    const float* __restrict__ src, _Float16* __restrict__ dst, int n4) {
  const int i = blockIdx.x * 256 + threadIdx.x;
  if (i >= n4) return;
  v4f x = ((const v4f*)src)[i];
  v4h y;
#pragma unroll
  for (int j = 0; j < 4; ++j) y[j] = (_Float16)x[j];
  ((v4h*)dst)[i] = y;
}

// ---------------------------------------------------------------------------
// K1: fused QKV projection GEMM, all-f16 operands, software-pipelined.
// One wave = 16x64 tile. 1536 output columns; [0,512)->Q, [512,1024)->K,
// [1024,1536)->V. Q written pre-scaled by 1/8; V written transposed [H][64][N].
// ---------------------------------------------------------------------------
__global__ __launch_bounds__(256) void qkv_proj_kernel(
    const _Float16* __restrict__ xq, const _Float16* __restrict__ xk,
    const _Float16* __restrict__ xv, const _Float16* __restrict__ W,
    const float* __restrict__ bias,
    _Float16* __restrict__ qb, _Float16* __restrict__ kb, _Float16* __restrict__ vb) {
  const int lane = threadIdx.x & 31;
  const int warp = threadIdx.x >> 5;
  const int wid  = blockIdx.x * 8 + warp;           // 3072 wave tiles
  const int row0 = (wid & 127) * 16;                // N/16 = 128
  const int col0 = (wid >> 7) * 64;                 // 1536/64 = 24
  const int sel  = col0 >> 9;                       // 0=Q, 1=K, 2=V
  const _Float16* A = (sel == 0) ? xq : (sel == 1) ? xk : xv;

  const int m   = lane & 15;
  const int hik = (lane >> 4) * 8;                  // A K-offset / C row-offset
  const int bK0 = (lane >> 4) * 8;                  // B K-offset
  const _Float16* arow = A + (size_t)(row0 + m) * EDIM + hik;
  const _Float16* wrow[4];
#pragma unroll
  for (int t = 0; t < 4; ++t)
    wrow[t] = W + (size_t)(col0 + t * 16 + m) * EDIM + bK0;  // B[k][n] = W[n][k]

  v8f acc[4];
#pragma unroll
  for (int t = 0; t < 4; ++t) acc[t] = {};

  // Software pipeline: fragments for step kc are loaded one step ahead so
  // each WMMA consumes loads that have had a full iteration to complete.
  v16h af = afrag_h(arow);
  v16h bf[4];
#pragma unroll
  for (int t = 0; t < 4; ++t) bf[t] = bfrag_h(wrow[t]);

#pragma unroll 2
  for (int kc = 0; kc < EDIM - 32; kc += 32) {
    v16h afn = afrag_h(arow + kc + 32);
    v16h bfn[4];
#pragma unroll
    for (int t = 0; t < 4; ++t) bfn[t] = bfrag_h(wrow[t] + kc + 32);
#pragma unroll
    for (int t = 0; t < 4; ++t) acc[t] = WMMA_F32_F16(af, bf[t], acc[t]);
    af = afn;
#pragma unroll
    for (int t = 0; t < 4; ++t) bf[t] = bfn[t];
  }
#pragma unroll
  for (int t = 0; t < 4; ++t) acc[t] = WMMA_F32_F16(af, bf[t], acc[t]);

#pragma unroll
  for (int t = 0; t < 4; ++t) {
    const int gcol = col0 + t * 16 + m;
    const int e = gcol & 511;
    const int h = e >> 6, d = e & 63;
    const float bv = bias[gcol];
#pragma unroll
    for (int r = 0; r < 8; ++r) {
      const int n = row0 + r + hik;                 // C: VGPR r -> M = r (+8 hi half)
      const float v = acc[t][r] + bv;
      if (sel == 0)      qb[((size_t)(h * N_NODES + n)) * HD + d] = (_Float16)(v * 0.125f);
      else if (sel == 1) kb[((size_t)(h * N_NODES + n)) * HD + d] = (_Float16)v;
      else               vb[((size_t)(h * HD + d)) * N_NODES + n] = (_Float16)v;   // transposed
    }
  }
}

// ---------------------------------------------------------------------------
// K2: per-(head,node) edge-type scores  ts[h][n][t] = q_scaled . edge_embs_K[t]
// ---------------------------------------------------------------------------
__global__ __launch_bounds__(256) void type_score_kernel(
    const _Float16* __restrict__ qb, const float* __restrict__ eK,
    float* __restrict__ ts) {
  const int idx = blockIdx.x * 256 + threadIdx.x;   // h*N + n
  if (idx >= NH * N_NODES) return;
  const _Float16* qrow = qb + (size_t)idx * HD;
  float q[HD];
#pragma unroll
  for (int d = 0; d < HD; ++d) q[d] = (float)qrow[d];
#pragma unroll
  for (int t = 0; t < NTYPES; ++t) {
    float s = 0.f;
#pragma unroll
    for (int d = 0; d < HD; ++d) s += q[d] * eK[t * HD + d];
    ts[(size_t)idx * NTYPES + t] = s;
  }
}

// ---------------------------------------------------------------------------
// K3: fused flash-style attention with edge-type bias. One wave owns
// (head h, 16-row block). Edge-type scores for the wave's 16 rows are staged
// in LDS once (constant over the j-loop). Per 32-key tile: eid + K + V loads
// issued up front; S = Q.K^T (4 WMMAs); bias from LDS; online softmax with
// max-butterflies only — row sums come from one extra WMMA against an
// all-ones B fragment; P via LDS relayout; O += P.V (4 WMMAs).
// ---------------------------------------------------------------------------
__global__ __launch_bounds__(256) void attn_kernel(
    const _Float16* __restrict__ qb, const _Float16* __restrict__ kb,
    const _Float16* __restrict__ vb, const float* __restrict__ ts,
    const int* __restrict__ eid, _Float16* __restrict__ aout) {
  __shared__ alignas(16) _Float16 pbuf[8][16][32];  // per-wave P staging (C -> A layout)
  __shared__ float tsw[8][16][NTYPES];              // per-wave edge-type scores
  const int lane = threadIdx.x & 31;
  const int warp = threadIdx.x >> 5;
  const int wid  = blockIdx.x * 8 + warp;           // 1024 waves
  const int h    = wid >> 7;
  const int row0 = (wid & 127) * 16;

  const int m   = lane & 15;
  const int hik = (lane >> 4) * 8;
  const int bK0 = (lane >> 4) * 8;

  // Stage this wave's 16 rows of edge-type scores into LDS (j-loop invariant).
  {
    const float* src = ts + (size_t)(h * N_NODES + row0) * NTYPES;
    for (int idx = lane; idx < 16 * NTYPES; idx += 32)
      tsw[warp][idx / NTYPES][idx % NTYPES] = src[idx];
  }
  __builtin_amdgcn_wave_barrier();
  asm volatile("s_wait_dscnt 0" ::: "memory");      // wave-local LDS RAW fence

  const _Float16* qrow = qb + ((size_t)(h * N_NODES + row0 + m)) * HD + hik;
  const v16h qf0 = afrag_h(qrow);
  const v16h qf1 = afrag_h(qrow + 32);

  v16h ones;                                        // all-ones B for row sums
#pragma unroll
  for (int i = 0; i < 16; ++i) ones[i] = (_Float16)1.0f;

  const _Float16* kbase = kb + ((size_t)h * N_NODES + m) * HD + bK0;     // + j*HD
  const _Float16* vbase = vb + ((size_t)(h * HD + m)) * N_NODES + bK0;   // + t*16*N + j0

  v8f O[4];
#pragma unroll
  for (int t = 0; t < 4; ++t) O[t] = {};
  float mrow[8], lrow[8];
#pragma unroll
  for (int r = 0; r < 8; ++r) { mrow[r] = -3.0e38f; lrow[r] = 0.f; }

  for (int j0 = 0; j0 < N_NODES; j0 += 32) {
    if (j0 + 32 < N_NODES) {                        // global_prefetch next tiles
      __builtin_prefetch(kbase + (size_t)(j0 + 32) * HD, 0, 3);
      __builtin_prefetch(kbase + (size_t)(j0 + 48) * HD, 0, 3);
      __builtin_prefetch(vbase + j0 + 32, 0, 3);
    }

    // ---- all global traffic for this tile issued up front:
    //      16 eid dwords + 8x2 b128 K/V fragments.
    int id0[8], id1[8];
#pragma unroll
    for (int r = 0; r < 8; ++r) {
      const size_t erow = (size_t)(row0 + r + hik) * N_NODES + j0;
      id0[r] = eid[erow + m];
      id1[r] = eid[erow + 16 + m];
    }
    const _Float16* k0 = kbase + (size_t)j0 * HD;
    const _Float16* k1 = k0 + 16 * HD;
    v16h kf0a = bfrag_h(k0), kf0b = bfrag_h(k0 + 32);
    v16h kf1a = bfrag_h(k1), kf1b = bfrag_h(k1 + 32);
    v16h vf[4];
#pragma unroll
    for (int t = 0; t < 4; ++t) vf[t] = bfrag_h(vbase + (size_t)t * 16 * N_NODES + j0);

    // ---- S = Q (16x64) @ K^T (64x32).
    v8f S0 = {}, S1 = {};
    S0 = WMMA_F32_F16(qf0, kf0a, S0);
    S1 = WMMA_F32_F16(qf0, kf1a, S1);
    S0 = WMMA_F32_F16(qf1, kf0b, S0);
    S1 = WMMA_F32_F16(qf1, kf1b, S1);

    // ---- LDS bias lookup + online softmax (max butterflies only).
    float alpha[8];
#pragma unroll
    for (int r = 0; r < 8; ++r) {
      float s0 = S0[r] + tsw[warp][r + hik][id0[r]];
      float s1 = S1[r] + tsw[warp][r + hik][id1[r]];
      const float mx   = half_max(fmaxf(s0, s1));
      const float mnew = fmaxf(mrow[r], mx);
      alpha[r] = __expf(mrow[r] - mnew);
      mrow[r]  = mnew;
      const float p0 = __expf(s0 - mnew);
      const float p1 = __expf(s1 - mnew);
#pragma unroll
      for (int t = 0; t < 4; ++t) O[t][r] *= alpha[r];
      pbuf[warp][r + hik][m]      = (_Float16)p0;   // C layout -> row-major 16x32
      pbuf[warp][r + hik][16 + m] = (_Float16)p1;
    }
    __builtin_amdgcn_wave_barrier();
    asm volatile("s_wait_dscnt 0" ::: "memory");    // wave-local LDS RAW fence

    // ---- O (16x64) += P (16x32) @ V (32x64); row sums via WMMA vs ones.
    v16h pf = afrag_h(&pbuf[warp][m][0] + hik);
    v8f rs = {};
    rs = WMMA_F32_F16(pf, ones, rs);                // rs[r] = rowsum(P[row r])
#pragma unroll
    for (int t = 0; t < 4; ++t) O[t] = WMMA_F32_F16(pf, vf[t], O[t]);
#pragma unroll
    for (int r = 0; r < 8; ++r) lrow[r] = lrow[r] * alpha[r] + rs[r];
    __builtin_amdgcn_wave_barrier();
  }

  // ---- normalize and emit f16 attention output in (N, E) layout.
#pragma unroll
  for (int r = 0; r < 8; ++r) {
    const int n = row0 + r + hik;
    const float inv = 1.0f / lrow[r];
#pragma unroll
    for (int t = 0; t < 4; ++t)
      aout[(size_t)n * EDIM + h * HD + t * 16 + m] = (_Float16)(O[t][r] * inv);
  }
}

// ---------------------------------------------------------------------------
// K4: output projection, all-f16 operands, software-pipelined, fp32 out.
// ---------------------------------------------------------------------------
__global__ __launch_bounds__(256) void out_proj_kernel(
    const _Float16* __restrict__ aout, const _Float16* __restrict__ W,
    const float* __restrict__ bias, float* __restrict__ out) {
  const int lane = threadIdx.x & 31;
  const int warp = threadIdx.x >> 5;
  const int wid  = blockIdx.x * 8 + warp;           // 1024 wave tiles
  const int row0 = (wid & 127) * 16;
  const int col0 = (wid >> 7) * 64;                 // 512/64 = 8 col tiles

  const int m   = lane & 15;
  const int hik = (lane >> 4) * 8;
  const int bK0 = (lane >> 4) * 8;
  const _Float16* arow = aout + (size_t)(row0 + m) * EDIM + hik;
  const _Float16* wrow[4];
#pragma unroll
  for (int t = 0; t < 4; ++t)
    wrow[t] = W + (size_t)(col0 + t * 16 + m) * EDIM + bK0;

  v8f acc[4];
#pragma unroll
  for (int t = 0; t < 4; ++t) acc[t] = {};

  v16h af = afrag_h(arow);
  v16h bf[4];
#pragma unroll
  for (int t = 0; t < 4; ++t) bf[t] = bfrag_h(wrow[t]);

#pragma unroll 2
  for (int kc = 0; kc < EDIM - 32; kc += 32) {
    v16h afn = afrag_h(arow + kc + 32);
    v16h bfn[4];
#pragma unroll
    for (int t = 0; t < 4; ++t) bfn[t] = bfrag_h(wrow[t] + kc + 32);
#pragma unroll
    for (int t = 0; t < 4; ++t) acc[t] = WMMA_F32_F16(af, bf[t], acc[t]);
    af = afn;
#pragma unroll
    for (int t = 0; t < 4; ++t) bf[t] = bfn[t];
  }
#pragma unroll
  for (int t = 0; t < 4; ++t) acc[t] = WMMA_F32_F16(af, bf[t], acc[t]);

#pragma unroll
  for (int t = 0; t < 4; ++t) {
    const int gcol = col0 + t * 16 + m;
    const float bv = bias[gcol];
#pragma unroll
    for (int r = 0; r < 8; ++r)
      out[(size_t)(row0 + r + hik) * EDIM + gcol] = acc[t][r] + bv;
  }
}

// ---------------------------------------------------------------------------
extern "C" void kernel_launch(void* const* d_in, const int* in_sizes, int n_in,
                              void* d_out, int out_size, void* d_ws, size_t ws_size,
                              hipStream_t stream) {
  const float* query = (const float*)d_in[0];
  const float* keym  = (const float*)d_in[1];
  const float* valm  = (const float*)d_in[2];
  const int*   eid   = (const int*)d_in[3];
  const float* in_w  = (const float*)d_in[4];   // (1536, 512)
  const float* in_b  = (const float*)d_in[5];   // (1536,)
  const float* out_w = (const float*)d_in[6];   // (512, 512)
  const float* out_b = (const float*)d_in[7];   // (512,)
  const float* eK    = (const float*)d_in[8];   // (10, 64)
  float* out = (float*)d_out;

  // Workspace layout (f16 buffers first, then fp32 type scores): ~17.4 MB.
  const size_t NE = (size_t)N_NODES * EDIM;                 // 1,048,576
  _Float16* qb   = (_Float16*)d_ws;                         // [H][N][64], pre-scaled
  _Float16* kb   = qb + NE;                                 // [H][N][64]
  _Float16* vb   = kb + NE;                                 // [H][64][N] (transposed)
  _Float16* aout = vb + NE;                                 // [N][E]
  _Float16* xq   = aout + NE;                               // f16 copies of inputs
  _Float16* xk   = xq + NE;
  _Float16* xv   = xk + NE;
  _Float16* w16  = xv + NE;                                 // (1536, 512) f16
  _Float16* wo16 = w16 + (size_t)3 * EDIM * EDIM;           // (512, 512) f16
  float*    ts   = (float*)(wo16 + (size_t)EDIM * EDIM);    // [H][N][10]

  dim3 blk(256);
  // K0: one-time f32 -> f16 conversions (bandwidth-trivial).
  cvt_f16_kernel<<<dim3((int)(NE / 4 / 256)), blk, 0, stream>>>(query, xq, (int)(NE / 4));
  cvt_f16_kernel<<<dim3((int)(NE / 4 / 256)), blk, 0, stream>>>(keym, xk, (int)(NE / 4));
  cvt_f16_kernel<<<dim3((int)(NE / 4 / 256)), blk, 0, stream>>>(valm, xv, (int)(NE / 4));
  cvt_f16_kernel<<<dim3(3 * EDIM * EDIM / 4 / 256), blk, 0, stream>>>(in_w, w16, 3 * EDIM * EDIM / 4);
  cvt_f16_kernel<<<dim3(EDIM * EDIM / 4 / 256), blk, 0, stream>>>(out_w, wo16, EDIM * EDIM / 4);
  // K1: 128 row-tiles x 24 col-tiles = 3072 waves / 8 per block
  qkv_proj_kernel<<<dim3(384), blk, 0, stream>>>(xq, xk, xv, w16, in_b, qb, kb, vb);
  // K2: 16384 (h,n) threads
  type_score_kernel<<<dim3((NH * N_NODES + 255) / 256), blk, 0, stream>>>(qb, eK, ts);
  // K3: 8 heads x 128 row-blocks = 1024 waves / 8 per block
  attn_kernel<<<dim3(128), blk, 0, stream>>>(qb, kb, vb, ts, eid, aout);
  // K4: 128 row-tiles x 8 col-tiles = 1024 waves / 8 per block
  out_proj_kernel<<<dim3(128), blk, 0, stream>>>(aout, wo16, out_b, out);
}
